// GINEBlock_68332929679679
// MI455X (gfx1250) — matrix operand order repaired and checked by reading, not compiled, not run on hardware
//
#include <hip/hip_runtime.h>
#include <hip/hip_bf16.h>

typedef __attribute__((ext_vector_type(16))) _Float16 v16h;
typedef __attribute__((ext_vector_type(8)))  _Float16 v8h;
typedef __attribute__((ext_vector_type(8)))  float    v8f;

#define N_NODES   50000
#define N_EDGES   800000
#define IN_DIM    96
#define OUT_DIM   96
#define EDGE_DIM  32
#define BN_EPS    1e-5f
#define N_NODE_TILES 3125

// ---------------------------------------------------------------------------
// Kernel 0: zero the scatter target and BN accumulators (float4 stores)
// ---------------------------------------------------------------------------
__global__ void __launch_bounds__(256) gine_zero_kernel(float4* __restrict__ aggr4,
                                                        float*  __restrict__ sums) {
    int idx = blockIdx.x * 256 + threadIdx.x;
    if (idx < (N_NODES * IN_DIM) / 4) aggr4[idx] = make_float4(0.f, 0.f, 0.f, 0.f);
    if (idx < 2 * OUT_DIM)            sums[idx] = 0.0f;
}

// ---------------------------------------------------------------------------
// Kernel 1: edge projection GEMM (WMMA) + message + atomic scatter-add
//   msg = relu(x[src] + edge_attr @ We + be);  aggr[dst] += msg
// One wave = one 16-edge tile. x[src] rows staged to LDS via b128 loads.
// All global accesses use unsigned 32-bit indices off uniform bases so the
// backend can emit SADDR(base SGPR pair) + 32-bit VGPR offset addressing.
// ---------------------------------------------------------------------------
__global__ void __launch_bounds__(256) gine_edge_kernel(
    const float* __restrict__ x,
    const int*   __restrict__ edge_index,   // [2, E]
    const float* __restrict__ edge_attr,    // [E, 32]
    const float* __restrict__ We,           // [32, 96]
    const float* __restrict__ be,           // [96]
    float*       __restrict__ aggr)         // [N, 96]
{
    __shared__ _Float16 sWeT[IN_DIM][EDGE_DIM];  // [n][k] transposed, 6 KB
    __shared__ int      sSrc[8][16];
    __shared__ unsigned sDoff[8][16];            // dst * 96
    __shared__ float    sX[8][16][IN_DIM];       // gathered x rows, 48 KB

    const int t    = threadIdx.x;
    const int wave = t >> 5;
    const int lane = t & 31;

    // Cooperative load of We^T into LDS (f32 -> f16)
    for (int i = t; i < IN_DIM * EDGE_DIM; i += 256) {
        int n = i >> 5, k = i & 31;
        sWeT[n][k] = (_Float16)We[(unsigned)(k * IN_DIM + n)];
    }

    const int tile  = blockIdx.x * 8 + wave;   // exact: 50000 tiles
    const int ebase = tile * 16;
    if (lane < 16) {
        sSrc[wave][lane]  = edge_index[(unsigned)(ebase + lane)];
        sDoff[wave][lane] = (unsigned)edge_index[(unsigned)(N_EDGES + ebase + lane)] * IN_DIM;
    }
    __syncthreads();

    // Stage gathered x[src] rows into LDS: 16 rows x 24 float4 = 384 tasks
#pragma unroll
    for (int j = 0; j < 12; ++j) {
        const int task = lane + 32 * j;
        const int row  = task / 24;
        const int ch   = task % 24;
        const unsigned sidx = (unsigned)sSrc[wave][row] * IN_DIM + ch * 4;
        const float4 v = *(const float4*)(x + sidx);
        *(float4*)(&sX[wave][row][ch * 4]) = v;
    }

    // A fragment (edge_attr 16x32 f16) via 4x float4 loads (u32 indexing)
    const int rowm = lane & 15;
    const int kb0  = (lane < 16) ? 0 : 8;
    const unsigned abase = (unsigned)(ebase + rowm) * EDGE_DIM + kb0;
    const float4 a0 = *(const float4*)(edge_attr + abase);
    const float4 a1 = *(const float4*)(edge_attr + abase + 4);
    const float4 a2 = *(const float4*)(edge_attr + abase + 16);
    const float4 a3 = *(const float4*)(edge_attr + abase + 20);
    v16h A;
    A[0]  = (_Float16)a0.x; A[1]  = (_Float16)a0.y; A[2]  = (_Float16)a0.z; A[3]  = (_Float16)a0.w;
    A[4]  = (_Float16)a1.x; A[5]  = (_Float16)a1.y; A[6]  = (_Float16)a1.z; A[7]  = (_Float16)a1.w;
    A[8]  = (_Float16)a2.x; A[9]  = (_Float16)a2.y; A[10] = (_Float16)a2.z; A[11] = (_Float16)a2.w;
    A[12] = (_Float16)a3.x; A[13] = (_Float16)a3.y; A[14] = (_Float16)a3.z; A[15] = (_Float16)a3.w;

    const int nlo  = lane & 15;
    const int kbB  = (lane < 16) ? 0 : 8;
    const int mAdd = (lane >= 16) ? 8 : 0;

    // Batched pre-loads: biases and unsigned 32-bit destination offsets
    float bias[6];
#pragma unroll
    for (int nt = 0; nt < 6; ++nt) bias[nt] = be[(unsigned)(nt * 16 + nlo)];
    unsigned po[8];
#pragma unroll
    for (int r = 0; r < 8; ++r) po[r] = sDoff[wave][r + mAdd] + (unsigned)nlo;

#pragma unroll
    for (int nt = 0; nt < 6; ++nt) {
        const int n = nt * 16 + nlo;
        const v8h* wrow = (const v8h*)&sWeT[n][0];     // 4 chunks of 8 halves
        const v8h blo = wrow[kbB >> 3];
        const v8h bhi = wrow[(kbB >> 3) + 2];
        v16h B;
#pragma unroll
        for (int h = 0; h < 8; ++h) { B[h] = blo[h]; B[8 + h] = bhi[h]; }

        // Fold the bias into the accumulator (off the post-WMMA critical path)
        v8f C;
#pragma unroll
        for (int r = 0; r < 8; ++r) C[r] = bias[nt];
        C = __builtin_amdgcn_wmma_f32_16x16x32_f16(false, A, false, B,
                                                   (short)0, C, false, false);
        // Batched LDS gathers, then independent atomics (saddr + u32 offset)
        float xv[8];
#pragma unroll
        for (int r = 0; r < 8; ++r) xv[r] = sX[wave][r + mAdd][n];
#pragma unroll
        for (int r = 0; r < 8; ++r) {
            float v = C[r] + xv[r];
            v = v > 0.0f ? v : 0.0f;
            atomicAdd(&aggr[po[r] + (unsigned)(nt * 16)], v);
        }
    }
}

// ---------------------------------------------------------------------------
// Kernel 2: node MLP: h = x + aggr; t = relu(h@W1+b1); h2 = t@W2+b2
// plus hierarchical BN statistics (regs -> LDS ds_add_f32 -> global atomics)
// ---------------------------------------------------------------------------
__global__ void __launch_bounds__(256) gine_mlp_kernel(
    const float* __restrict__ x,
    const float* __restrict__ aggr,
    const float* __restrict__ W1,   // [96, 96]
    const float* __restrict__ b1,   // [96]
    const float* __restrict__ W2,   // [96, 96]
    const float* __restrict__ b2,   // [96]
    float*       __restrict__ h2,   // [N, 96]  (pre-BN activations)
    float*       __restrict__ gSum, // [96]
    float*       __restrict__ gSq)  // [96]
{
    __shared__ _Float16 sW1T[OUT_DIM][IN_DIM];     // 18 KB, [n][k]
    __shared__ _Float16 sW2T[OUT_DIM][OUT_DIM];    // 18 KB, [n][k]
    __shared__ _Float16 tbuf[8][16][OUT_DIM];      // 24 KB per-wave intermediates
    __shared__ float    sSum[OUT_DIM];
    __shared__ float    sSq[OUT_DIM];

    const int t    = threadIdx.x;
    const int wave = t >> 5;
    const int lane = t & 31;

    for (int i = t; i < OUT_DIM * IN_DIM; i += 256) {
        int n = i / IN_DIM, k = i % IN_DIM;
        sW1T[n][k] = (_Float16)W1[(unsigned)(k * OUT_DIM + n)];
        sW2T[n][k] = (_Float16)W2[(unsigned)(k * OUT_DIM + n)];
    }
    if (t < OUT_DIM) { sSum[t] = 0.0f; sSq[t] = 0.0f; }
    __syncthreads();

    const int  tile   = blockIdx.x * 8 + wave;
    const bool active = (tile < N_NODE_TILES);

    if (active) {
        const int nodebase = tile * 16;
        const int rowm = lane & 15;
        const int node = nodebase + rowm;
        const int kb0  = (lane < 16) ? 0 : 8;
        const int nlo  = lane & 15;
        const int kbB  = (lane < 16) ? 0 : 8;
        const int mAdd = (lane >= 16) ? 8 : 0;

        // Batched bias pre-loads
        float b1v[6], b2v[6];
#pragma unroll
        for (int nt = 0; nt < 6; ++nt) {
            b1v[nt] = b1[(unsigned)(nt * 16 + nlo)];
            b2v[nt] = b2[(unsigned)(nt * 16 + nlo)];
        }
        // Unsigned 32-bit h2 row offsets (saddr-form stores)
        unsigned ho[8];
#pragma unroll
        for (int r = 0; r < 8; ++r)
            ho[r] = (unsigned)(nodebase + r + mAdd) * OUT_DIM + (unsigned)nlo;

        // A fragments of h = x + aggr via float4 loads (u32 indexing)
        const unsigned roff = (unsigned)node * IN_DIM;
        v16h A1[3];
#pragma unroll
        for (int kt = 0; kt < 3; ++kt) {
            const unsigned cb = roff + (unsigned)(kt * 32 + kb0);
            const float4 x0 = *(const float4*)(x + cb);
            const float4 x1 = *(const float4*)(x + cb + 4);
            const float4 x2 = *(const float4*)(x + cb + 16);
            const float4 x3 = *(const float4*)(x + cb + 20);
            const float4 g0 = *(const float4*)(aggr + cb);
            const float4 g1 = *(const float4*)(aggr + cb + 4);
            const float4 g2 = *(const float4*)(aggr + cb + 16);
            const float4 g3 = *(const float4*)(aggr + cb + 20);
            A1[kt][0]  = (_Float16)(x0.x + g0.x); A1[kt][1]  = (_Float16)(x0.y + g0.y);
            A1[kt][2]  = (_Float16)(x0.z + g0.z); A1[kt][3]  = (_Float16)(x0.w + g0.w);
            A1[kt][4]  = (_Float16)(x1.x + g1.x); A1[kt][5]  = (_Float16)(x1.y + g1.y);
            A1[kt][6]  = (_Float16)(x1.z + g1.z); A1[kt][7]  = (_Float16)(x1.w + g1.w);
            A1[kt][8]  = (_Float16)(x2.x + g2.x); A1[kt][9]  = (_Float16)(x2.y + g2.y);
            A1[kt][10] = (_Float16)(x2.z + g2.z); A1[kt][11] = (_Float16)(x2.w + g2.w);
            A1[kt][12] = (_Float16)(x3.x + g3.x); A1[kt][13] = (_Float16)(x3.y + g3.y);
            A1[kt][14] = (_Float16)(x3.z + g3.z); A1[kt][15] = (_Float16)(x3.w + g3.w);
        }

        // Stage 1: t = relu(h @ W1 + b1) -> LDS (f16)
#pragma unroll
        for (int ntile = 0; ntile < 6; ++ntile) {
            const int n = ntile * 16 + nlo;
            const v8h* w1row = (const v8h*)&sW1T[n][0];
            v8f C;
#pragma unroll
            for (int r = 0; r < 8; ++r) C[r] = b1v[ntile];
#pragma unroll
            for (int kt = 0; kt < 3; ++kt) {
                const int cb = (kt * 32 + kbB) >> 3;
                const v8h blo = w1row[cb], bhi = w1row[cb + 2];
                v16h B;
#pragma unroll
                for (int h = 0; h < 8; ++h) { B[h] = blo[h]; B[8 + h] = bhi[h]; }
                C = __builtin_amdgcn_wmma_f32_16x16x32_f16(false, A1[kt], false, B,
                                                           (short)0, C, false, false);
            }
#pragma unroll
            for (int r = 0; r < 8; ++r) {
                float v = C[r];
                v = v > 0.0f ? v : 0.0f;
                tbuf[wave][r + mAdd][n] = (_Float16)v;
            }
        }

        // Stage 2: h2 = t @ W2 + b2, plus per-column BN partial sums
        const v8h* trow = (const v8h*)&tbuf[wave][rowm][0];  // 12 chunks of 8
        v16h A2[3];
#pragma unroll
        for (int kt = 0; kt < 3; ++kt) {
            const int cb = (kt * 32 + kb0) >> 3;
            const v8h lo = trow[cb], hi = trow[cb + 2];
#pragma unroll
            for (int h = 0; h < 8; ++h) { A2[kt][h] = lo[h]; A2[kt][8 + h] = hi[h]; }
        }

#pragma unroll
        for (int ntile = 0; ntile < 6; ++ntile) {
            const int n = ntile * 16 + nlo;
            const v8h* w2row = (const v8h*)&sW2T[n][0];
            v8f C;
#pragma unroll
            for (int r = 0; r < 8; ++r) C[r] = b2v[ntile];
#pragma unroll
            for (int kt = 0; kt < 3; ++kt) {
                const int cb = (kt * 32 + kbB) >> 3;
                const v8h blo = w2row[cb], bhi = w2row[cb + 2];
                v16h B;
#pragma unroll
                for (int h = 0; h < 8; ++h) { B[h] = blo[h]; B[8 + h] = bhi[h]; }
                C = __builtin_amdgcn_wmma_f32_16x16x32_f16(false, A2[kt], false, B,
                                                           (short)0, C, false, false);
            }
            float lsum = 0.0f, lsq = 0.0f;
#pragma unroll
            for (int r = 0; r < 8; ++r) {
                const float v = C[r];
                h2[ho[r] + (unsigned)(ntile * 16)] = v;
                lsum += v;
                lsq  += v * v;
            }
            atomicAdd(&sSum[n], lsum);   // LDS ds_add_f32
            atomicAdd(&sSq[n],  lsq);
        }
    }

    __syncthreads();
    if (t < OUT_DIM) {
        atomicAdd(&gSum[t], sSum[t]);
        atomicAdd(&gSq[t],  sSq[t]);
    }
}

// ---------------------------------------------------------------------------
// Kernel 3: BatchNorm finalize + affine + ReLU (float4 per thread)
// ---------------------------------------------------------------------------
__global__ void __launch_bounds__(256) gine_bn_kernel(
    const float4* __restrict__ h24,
    const float*  __restrict__ gSum,
    const float*  __restrict__ gSq,
    const float*  __restrict__ gamma,
    const float*  __restrict__ beta,
    float4*       __restrict__ out4)
{
    const int idx4 = blockIdx.x * 256 + threadIdx.x;
    if (idx4 >= (N_NODES * OUT_DIM) / 4) return;
    const int c = (idx4 * 4) % OUT_DIM;        // 4 | 96 -> no wrap inside a float4
    const float invN = 1.0f / (float)N_NODES;
    const float4 hv = h24[(unsigned)idx4];
    float4 o;
    {
        const float m = gSum[c] * invN;
        const float s = rsqrtf(gSq[c] * invN - m * m + BN_EPS) * gamma[c];
        o.x = (hv.x - m) * s + beta[c];
    }
    {
        const float m = gSum[c + 1] * invN;
        const float s = rsqrtf(gSq[c + 1] * invN - m * m + BN_EPS) * gamma[c + 1];
        o.y = (hv.y - m) * s + beta[c + 1];
    }
    {
        const float m = gSum[c + 2] * invN;
        const float s = rsqrtf(gSq[c + 2] * invN - m * m + BN_EPS) * gamma[c + 2];
        o.z = (hv.z - m) * s + beta[c + 2];
    }
    {
        const float m = gSum[c + 3] * invN;
        const float s = rsqrtf(gSq[c + 3] * invN - m * m + BN_EPS) * gamma[c + 3];
        o.w = (hv.w - m) * s + beta[c + 3];
    }
    o.x = o.x > 0.0f ? o.x : 0.0f;
    o.y = o.y > 0.0f ? o.y : 0.0f;
    o.z = o.z > 0.0f ? o.z : 0.0f;
    o.w = o.w > 0.0f ? o.w : 0.0f;
    out4[(unsigned)idx4] = o;
}

// ---------------------------------------------------------------------------
// Host launcher
// ---------------------------------------------------------------------------
extern "C" void kernel_launch(void* const* d_in, const int* in_sizes, int n_in,
                              void* d_out, int out_size, void* d_ws, size_t ws_size,
                              hipStream_t stream) {
    const float* x          = (const float*)d_in[0];
    const int*   edge_index = (const int*)  d_in[1];
    const float* edge_attr  = (const float*)d_in[2];
    const float* We         = (const float*)d_in[3];
    const float* be         = (const float*)d_in[4];
    const float* W1         = (const float*)d_in[5];
    const float* b1         = (const float*)d_in[6];
    const float* W2         = (const float*)d_in[7];
    const float* b2         = (const float*)d_in[8];
    const float* gamma      = (const float*)d_in[9];
    const float* beta       = (const float*)d_in[10];

    // Workspace layout
    float* aggr = (float*)d_ws;                        // [N, 96]
    float* h2   = aggr + (size_t)N_NODES * IN_DIM;     // [N, 96]
    float* gSum = h2   + (size_t)N_NODES * OUT_DIM;    // [96] sum, [96] sumsq
    float* gSq  = gSum + OUT_DIM;

    const int elems4 = (N_NODES * IN_DIM) / 4;         // 1.2M float4
    const int vblk   = (elems4 + 255) / 256;

    gine_zero_kernel<<<vblk, 256, 0, stream>>>((float4*)aggr, gSum);

    // 800000 edges / 16 per wave / 8 waves per block = 6250 blocks (exact)
    gine_edge_kernel<<<6250, 256, 0, stream>>>(x, edge_index, edge_attr, We, be, aggr);

    // 3125 node tiles / 8 waves per block = 391 blocks (last partially active)
    gine_mlp_kernel<<<(N_NODE_TILES + 7) / 8, 256, 0, stream>>>(
        x, aggr, W1, b1, W2, b2, h2, gSum, gSq);

    gine_bn_kernel<<<vblk, 256, 0, stream>>>((const float4*)h2, gSum, gSq,
                                             gamma, beta, (float4*)d_out);
}
